// Grid2Mesh_Encoder_62388694942507
// MI455X (gfx1250) — matrix operand order repaired and checked by reading, not compiled, not run on hardware
//
#include <hip/hip_runtime.h>
#include <hip/hip_bf16.h>
#include <math.h>

// ---------------- problem constants (from setup_inputs) ----------------
#define GH      181
#define GW      360
#define N_GRID  (GH * GW)        // 65160
#define NMESH   40962
#define NTOT    (N_GRID + NMESH) // 106122
#define C_IN    69
#define KPAD    96               // C_IN padded to multiple of 32 for WMMA
#define HID     256

typedef __bf16 v16bf __attribute__((ext_vector_type(16)));
typedef float  v8f   __attribute__((ext_vector_type(8)));

__device__ __forceinline__ float gelu_exact(float v) {
    return 0.5f * v * (1.0f + erff(v * 0.7071067811865476f));
}

// ---------------------------------------------------------------------
// h0 build: grid part = transpose of x (C,HW) -> (node, KPAD) bf16, zero pad
// ---------------------------------------------------------------------
__global__ void build_h0_grid(const float* __restrict__ x, __bf16* __restrict__ h0) {
    __shared__ float tile[32][33];
    int n0 = blockIdx.x * 32;
    int c0 = blockIdx.y * 32;
    int tx = threadIdx.x, ty = threadIdx.y;   // 32 x 8
    #pragma unroll
    for (int i = 0; i < 32; i += 8) {
        int c = c0 + ty + i;
        int n = n0 + tx;
        float v = 0.0f;
        if (c < C_IN && n < N_GRID) v = x[(size_t)c * N_GRID + n];
        tile[ty + i][tx] = v;
    }
    __syncthreads();
    #pragma unroll
    for (int i = 0; i < 32; i += 8) {
        int n = n0 + ty + i;
        int c = c0 + tx;
        if (n < N_GRID && c < KPAD) h0[(size_t)n * KPAD + c] = (__bf16)tile[tx][ty + i];
    }
}

// ---------------------------------------------------------------------
// h0 build: mesh part = bilinear x4 upsample gather (+ pole row)
// ---------------------------------------------------------------------
__global__ void build_h0_mesh(const float* __restrict__ x,
                              const int* __restrict__ Lat,
                              const int* __restrict__ Lon,
                              __bf16* __restrict__ h0) {
    size_t idx = (size_t)blockIdx.x * blockDim.x + threadIdx.x;
    if (idx >= (size_t)NMESH * KPAD) return;
    int c = (int)(idx % KPAD);
    int m = (int)(idx / KPAD);
    float v = 0.0f;
    if (c < C_IN) {
        int r = Lat[m] - 1;   // 0..720 (720 == pole row)
        int q = Lon[m] - 1;   // 0..1439
        const float* xc = x + (size_t)c * (GH * GW);
        if (r == 720) {
            v = xc[180 * GW + 0];                    // x[:, :, -1, 0]
        } else {
            float sh = fmaxf((r + 0.5f) * 0.25f - 0.5f, 0.0f);
            int i0 = (int)floorf(sh); i0 = max(0, min(i0, 179));
            int i1 = min(i0 + 1, 179);
            float wh = sh - (float)i0;
            float sw = fmaxf((q + 0.5f) * 0.25f - 0.5f, 0.0f);
            int j0 = (int)floorf(sw); j0 = max(0, min(j0, 359));
            int j1 = min(j0 + 1, 359);
            float ww = sw - (float)j0;
            float a00 = xc[i0 * GW + j0], a01 = xc[i0 * GW + j1];
            float a10 = xc[i1 * GW + j0], a11 = xc[i1 * GW + j1];
            v = (1.0f - wh) * ((1.0f - ww) * a00 + ww * a01)
              +          wh * ((1.0f - ww) * a10 + ww * a11);
        }
    }
    h0[(size_t)(N_GRID + m) * KPAD + c] = (__bf16)v;
}

// ---------------------------------------------------------------------
// degree / symmetric norm
// ---------------------------------------------------------------------
__global__ void deg_init(float* __restrict__ deg) {
    int i = blockIdx.x * blockDim.x + threadIdx.x;
    if (i < NTOT) deg[i] = 1.0f;   // self loop
}
__global__ void deg_accum(const int* __restrict__ ei, int E, float* __restrict__ deg) {
    int e = blockIdx.x * blockDim.x + threadIdx.x;
    if (e < E) atomicAdd(&deg[ei[E + e]], 1.0f);   // dst row
}
__global__ void dinv_finalize(float* __restrict__ deg) {
    int i = blockIdx.x * blockDim.x + threadIdx.x;
    if (i < NTOT) deg[i] = __frsqrt_rn(deg[i]);    // deg >= 1 always
}

// ---------------------------------------------------------------------
// Weight packer: W[K x 256] f32 -> fragment-swizzled bf16
// layout: [ntile(16)][kchunk(nkc)][lane(32)][elem(16)], zero pad k >= Kb
// lane n-col = lane&15 ; lane k = (lane>>4)*8 + (j&7) + (j>>3)*16
// ---------------------------------------------------------------------
__global__ void pack_w(const float* __restrict__ W, int Kb, int nkc,
                       __bf16* __restrict__ Bp) {
    int idx = blockIdx.x * blockDim.x + threadIdx.x;
    int total = 16 * nkc * 512;
    if (idx >= total) return;
    int j    = idx & 15;
    int lane = (idx >> 4) & 31;
    int t    = idx >> 9;
    int kc   = t % nkc;
    int nt   = t / nkc;
    int n = nt * 16 + (lane & 15);
    int k = kc * 32 + ((lane >> 4) << 3) + (j & 7) + ((j >> 3) << 4);
    Bp[idx] = (k < Kb) ? (__bf16)W[(size_t)k * HID + n] : (__bf16)0.0f;
}

// ---------------------------------------------------------------------
// WMMA GEMM:  D[M x 256] = A[M x K](bf16) * B(packed bf16 fragments)
// block = 4 waves; wave w computes a 16x64 strip (4 n-tiles), one A frag
// feeds 4 WMMAs per k-chunk. grid.x = ceil(M/16).
// MODE 0: f32   MODE 1: bf16(x+bias)   MODE 2: f32 gelu(x+bias)
// ---------------------------------------------------------------------
template <int MODE>
__global__ void wmma_gemm(const __bf16* __restrict__ A, int lda,
                          const __bf16* __restrict__ Bp, int nkc,
                          const float* __restrict__ bias,
                          void* __restrict__ Dst, int M) {
    const int lane = threadIdx.x & 31;
    const int wave = threadIdx.x >> 5;
    const int m0 = blockIdx.x * 16;

    // CDNA5 bf16 A fragment striping: lanes 0-15 K halves {0..7,16..23},
    // lanes 16-31 K halves {8..15,24..31}; rows = lane&15.
    const int am  = lane & 15;
    const int akb = (lane >> 4) << 3;    // 0 or 8
    int arow = m0 + am;
    if (arow >= M) arow = M - 1;         // clamp; stores are guarded
    const __bf16* ap = A + (size_t)arow * lda + akb;

    v8f acc[4] = {};
    for (int kc = 0; kc < nkc; ++kc) {
        const __bf16* ak = ap + kc * 32;
        __builtin_prefetch(ak + 32, 0, 3);         // speculative; OOB dropped
        v16bf a;
        #pragma unroll
        for (int i = 0; i < 8; ++i) {
            a[i]     = ak[i];                      // 16B contiguous
            a[8 + i] = ak[16 + i];                 // 16B contiguous
        }
        #pragma unroll
        for (int nt = 0; nt < 4; ++nt) {
            int ntile = wave * 4 + nt;
            const v16bf b = *(const v16bf*)(Bp +
                ((((size_t)ntile * nkc + kc) * 32 + lane) << 4));
            acc[nt] = __builtin_amdgcn_wmma_f32_16x16x32_bf16(
                          false, a, false, b, (short)0, acc[nt], false, false);
        }
    }

    const int rbase = m0 + ((lane >> 4) << 3);
    #pragma unroll
    for (int nt = 0; nt < 4; ++nt) {
        const int col = wave * 64 + nt * 16 + (lane & 15);
        if (m0 + 16 <= M) {                        // fast path: whole tile in range
            #pragma unroll
            for (int r = 0; r < 8; ++r) {
                int row = rbase + r;
                float v = acc[nt][r];
                if constexpr (MODE == 0)
                    ((float*)Dst)[(size_t)row * HID + col] = v;
                else if constexpr (MODE == 1)
                    ((__bf16*)Dst)[(size_t)row * HID + col] = (__bf16)(v + bias[col]);
                else
                    ((float*)Dst)[(size_t)row * HID + col] = gelu_exact(v + bias[col]);
            }
        } else {
            #pragma unroll
            for (int r = 0; r < 8; ++r) {
                int row = rbase + r;
                if (row < M) {
                    float v = acc[nt][r];
                    if constexpr (MODE == 0)
                        ((float*)Dst)[(size_t)row * HID + col] = v;
                    else if constexpr (MODE == 1)
                        ((__bf16*)Dst)[(size_t)row * HID + col] = (__bf16)(v + bias[col]);
                    else
                        ((float*)Dst)[(size_t)row * HID + col] = gelu_exact(v + bias[col]);
                }
            }
        }
    }
}

// ---------------------------------------------------------------------
// GCN aggregation
// ---------------------------------------------------------------------
__global__ void agg_init(const float* __restrict__ dinv,
                         const float* __restrict__ Y,
                         float* __restrict__ H) {
    size_t i = (size_t)blockIdx.x * blockDim.x + threadIdx.x;
    if (i < (size_t)NTOT * HID) {
        float d = dinv[i >> 8];
        H[i] = d * d * Y[i];     // self-loop contribution
    }
}
__global__ void agg_edges(const int* __restrict__ ei, int E,
                          const float* __restrict__ dinv,
                          const float* __restrict__ Y,
                          float* __restrict__ H) {
    int t = blockIdx.x * blockDim.x + threadIdx.x;
    int e = t >> 6;              // 64 lanes per edge, 4 floats per lane
    if (e >= E) return;
    int l = (t & 63) << 2;
    int s = ei[e];
    int d = ei[E + e];
    float nm = dinv[s] * dinv[d];
    const float4 y = *(const float4*)(Y + (size_t)s * HID + l);
    float* hp = H + (size_t)d * HID + l;
    atomicAdd(hp + 0, nm * y.x);
    atomicAdd(hp + 1, nm * y.y);
    atomicAdd(hp + 2, nm * y.z);
    atomicAdd(hp + 3, nm * y.w);
}
__global__ void post_gcn1(const float* __restrict__ H,
                          const float* __restrict__ b1,
                          __bf16* __restrict__ G) {
    size_t i = (size_t)blockIdx.x * blockDim.x + threadIdx.x;
    if (i < (size_t)NTOT * HID)
        G[i] = (__bf16)gelu_exact(H[i] + b1[i & (HID - 1)]);
}
__global__ void mesh_extract(const float* __restrict__ H,
                             const float* __restrict__ b2,
                             __bf16* __restrict__ Mb) {
    size_t i = (size_t)blockIdx.x * blockDim.x + threadIdx.x;
    if (i < (size_t)NMESH * HID)
        Mb[i] = (__bf16)(H[(size_t)N_GRID * HID + i] + b2[i & (HID - 1)]);
}

// ---------------------------------------------------------------------
// (NMESH x 256) -> (256 x NMESH) tiled transpose to d_out
// ---------------------------------------------------------------------
__global__ void transpose_out(const float* __restrict__ T, float* __restrict__ out) {
    __shared__ float tile[32][33];
    int m0 = blockIdx.x * 32;
    int c0 = blockIdx.y * 32;
    #pragma unroll
    for (int i = threadIdx.y; i < 32; i += 8) {
        int m = m0 + i;
        int c = c0 + threadIdx.x;
        tile[i][threadIdx.x] = (m < NMESH) ? T[(size_t)m * HID + c] : 0.0f;
    }
    __syncthreads();
    #pragma unroll
    for (int i = threadIdx.y; i < 32; i += 8) {
        int c = c0 + i;
        int m = m0 + threadIdx.x;
        if (m < NMESH) out[(size_t)c * NMESH + m] = tile[threadIdx.x][i];
    }
}

// ---------------------------------------------------------------------
extern "C" void kernel_launch(void* const* d_in, const int* in_sizes, int n_in,
                              void* d_out, int out_size, void* d_ws, size_t ws_size,
                              hipStream_t stream) {
    const float* x   = (const float*)d_in[0];
    const int*   ei  = (const int*)  d_in[1];
    const int*   Lat = (const int*)  d_in[2];
    const int*   Lon = (const int*)  d_in[3];
    const float* W1  = (const float*)d_in[4];
    const float* b1  = (const float*)d_in[5];
    const float* W2  = (const float*)d_in[6];
    const float* b2  = (const float*)d_in[7];
    const float* Wl  = (const float*)d_in[8];
    const float* bl  = (const float*)d_in[9];
    const float* Wl1 = (const float*)d_in[10];
    const float* bl1 = (const float*)d_in[11];
    const int E = in_sizes[1] / 2;

    // workspace carve-up
    char* ws = (char*)d_ws;
    size_t off = 0;
    auto alloc = [&](size_t bytes) {
        size_t p = off;
        off = (off + bytes + 255) & ~(size_t)255;
        return p;
    };
    float*  dinv = (float*)(ws + alloc((size_t)NTOT * 4));
    __bf16* h0   = (__bf16*)(ws + alloc((size_t)NTOT * KPAD * 2));
    float*  Y    = (float*)(ws + alloc((size_t)NTOT * HID * 4));
    float*  Hb   = (float*)(ws + alloc((size_t)NTOT * HID * 4));
    __bf16* Gbf  = (__bf16*)(ws + alloc((size_t)NTOT * HID * 2));
    __bf16* W1p  = (__bf16*)(ws + alloc((size_t)16 * 3 * 512 * 2));
    __bf16* W2p  = (__bf16*)(ws + alloc((size_t)16 * 8 * 512 * 2));
    __bf16* Wlp  = (__bf16*)(ws + alloc((size_t)16 * 8 * 512 * 2));
    __bf16* Wl1p = (__bf16*)(ws + alloc((size_t)16 * 8 * 512 * 2));
    __bf16* T1bf = (__bf16*)Y;   // reuse: Y dead after 2nd aggregation
    float*  Tout = Hb;           // reuse: H dead after mesh extraction

    const int Mtiles  = (NTOT + 15) / 16;    // 6633
    const int MtilesM = (NMESH + 15) / 16;   // 2561

    // 0) pack weights to fragment-swizzled bf16 (tiny, L2-resident)
    pack_w<<<(16 * 3 * 512 + 255) / 256, 256, 0, stream>>>(W1, C_IN, 3, W1p);
    pack_w<<<(16 * 8 * 512 + 255) / 256, 256, 0, stream>>>(W2, HID, 8, W2p);
    pack_w<<<(16 * 8 * 512 + 255) / 256, 256, 0, stream>>>(Wl, HID, 8, Wlp);
    pack_w<<<(16 * 8 * 512 + 255) / 256, 256, 0, stream>>>(Wl1, HID, 8, Wl1p);

    // 1) node features
    build_h0_grid<<<dim3((N_GRID + 31) / 32, 3), dim3(32, 8), 0, stream>>>(x, h0);
    {
        size_t n = (size_t)NMESH * KPAD;
        build_h0_mesh<<<(unsigned)((n + 255) / 256), 256, 0, stream>>>(x, Lat, Lon, h0);
    }

    // 2) symmetric GCN norm
    deg_init<<<(NTOT + 255) / 256, 256, 0, stream>>>(dinv);
    deg_accum<<<(E + 255) / 256, 256, 0, stream>>>(ei, E, dinv);
    dinv_finalize<<<(NTOT + 255) / 256, 256, 0, stream>>>(dinv);

    const size_t NH = (size_t)NTOT * HID;
    const unsigned aggBlocks = (unsigned)(((size_t)E * 64 + 255) / 256);

    // 3) GCN layer 1
    wmma_gemm<0><<<Mtiles, 128, 0, stream>>>(h0, KPAD, W1p, 3, nullptr, (void*)Y, NTOT);
    agg_init<<<(unsigned)((NH + 255) / 256), 256, 0, stream>>>(dinv, Y, Hb);
    agg_edges<<<aggBlocks, 256, 0, stream>>>(ei, E, dinv, Y, Hb);
    post_gcn1<<<(unsigned)((NH + 255) / 256), 256, 0, stream>>>(Hb, b1, Gbf);

    // 4) GCN layer 2
    wmma_gemm<0><<<Mtiles, 128, 0, stream>>>(Gbf, HID, W2p, 8, nullptr, (void*)Y, NTOT);
    agg_init<<<(unsigned)((NH + 255) / 256), 256, 0, stream>>>(dinv, Y, Hb);
    agg_edges<<<aggBlocks, 256, 0, stream>>>(ei, E, dinv, Y, Hb);

    // 5) mesh MLP
    {
        size_t n = (size_t)NMESH * HID;
        mesh_extract<<<(unsigned)((n + 255) / 256), 256, 0, stream>>>(Hb, b2, Gbf);
    }
    wmma_gemm<1><<<MtilesM, 128, 0, stream>>>(Gbf, HID, Wlp, 8, bl, (void*)T1bf, NMESH);
    wmma_gemm<2><<<MtilesM, 128, 0, stream>>>(T1bf, HID, Wl1p, 8, bl1, (void*)Tout, NMESH);

    // 6) (NMESH,256) -> (256,NMESH)
    transpose_out<<<dim3((NMESH + 31) / 32, HID / 32), dim3(32, 8), 0, stream>>>(
        Tout, (float*)d_out);
}
